// UnetGCN_14242111553927
// MI455X (gfx1250) — matrix operand order repaired and checked by reading, not compiled, not run on hardware
//
#include <hip/hip_runtime.h>
#include <hip/hip_bf16.h>

// ---------------------------------------------------------------------------
// Graph U-Net (dense-adjacency GCN) for MI455X / gfx1250, wave32 + WMMA.
//   - A@A (augment_adj): exact-in-f16 integer adjacency -> v_wmma_f32_16x16x32_f16
//   - X@Theta and A^T@Y : fp32 -> v_wmma_f32_16x16x4_f32
//   - GEMM staging via global_load_async_to_lds_b128 + ping-pong LDS buffers
// ---------------------------------------------------------------------------

typedef __attribute__((ext_vector_type(2)))  float    v2f;
typedef __attribute__((ext_vector_type(8)))  float    v8f;
typedef __attribute__((ext_vector_type(16))) _Float16 v16h;
typedef __attribute__((ext_vector_type(2)))  _Float16 v2h;

#define N0 4096
#define N1 2048
#define N2 1024

#if defined(__HIP_DEVICE_COMPILE__) && __has_builtin(__builtin_amdgcn_wmma_f32_16x16x4_f32)
#define HAVE_WMMA_F32X4 1
#else
#define HAVE_WMMA_F32X4 0
#endif

#define AS_F 36    // feat-A LDS stride ([m][k], 32 k + pad, even -> 8B-aligned pair loads)
#define KS_S 132   // [k][m] / [k][n] LDS stride (128 + pad, multiple of 4 -> 16B stores)
#define AH_S 36    // f16 LDS stride in halves (even -> 4B-aligned pair loads)

// ---- CDNA5 async global->LDS copy (ASYNCcnt) -------------------------------
__device__ __forceinline__ void async_ld_b128(const float* g, float* lds) {
  unsigned lds_off = (unsigned)(uintptr_t)lds;           // low 32 bits = LDS offset
  unsigned long long ga = (unsigned long long)(uintptr_t)g;
  asm volatile("global_load_async_to_lds_b128 %0, %1, off"
               :: "v"(lds_off), "v"(ga) : "memory");
}
__device__ __forceinline__ void wait_async0() {
#if defined(__HIP_DEVICE_COMPILE__) && __has_builtin(__builtin_amdgcn_s_wait_asynccnt)
  __builtin_amdgcn_s_wait_asynccnt(0);
#else
  asm volatile("s_wait_asynccnt 0" ::: "memory");
#endif
}

// ---------------------------------------------------------------------------
// Small utility kernels
// ---------------------------------------------------------------------------

__global__ __launch_bounds__(256) void k_scatter_edges(const int* __restrict__ ei,
                                                       float* __restrict__ A, int n, int e) {
  int t = blockIdx.x * blockDim.x + threadIdx.x;
  if (t < e) {
    int s = ei[t];
    int d = ei[e + t];
    atomicAdd(&A[(long)s * n + d], 1.0f);
  }
}

__global__ __launch_bounds__(256) void k_add_eye(float* __restrict__ A, int n) {
  int i = blockIdx.x * blockDim.x + threadIdx.x;
  if (i < n) A[(long)i * n + i] += 1.0f;
}

// deg[i] = sum_j A[i][j] (A already contains +I); dinv = rsqrt(deg) (deg>0)
__global__ __launch_bounds__(256) void k_row_dinv(const float* __restrict__ A,
                                                  float* __restrict__ dinv, int n) {
  int wave = threadIdx.x >> 5;
  int lane = threadIdx.x & 31;
  int row = blockIdx.x * 8 + wave;
  if (row >= n) return;
  const float* r = A + (long)row * n;
  float s = 0.f;
  for (int j = lane; j < n; j += 32) s += r[j];
  #pragma unroll
  for (int o = 16; o > 0; o >>= 1) s += __shfl_xor(s, o, 32);
  if (lane == 0) dinv[row] = (s > 0.f) ? rsqrtf(s) : 0.f;
}

// ---------------------------------------------------------------------------
// GEMM 1: Y[r][c] = dinv[r] * sum_k X[r][k] * Th[k][c]   (fp32 WMMA 16x16x4)
// Block tile 128x128, 8 waves (4Mx2N), wave = 2x4 subtiles of 16x16.
// Async double-buffered LDS staging.
// ---------------------------------------------------------------------------
__global__ __launch_bounds__(256) void k_gemm_feat(const float* __restrict__ X,
                                                   const float* __restrict__ Th,
                                                   const float* __restrict__ dinv,
                                                   float* __restrict__ Y,
                                                   int n, int K, int co) {
  __shared__ float As[2][128 * AS_F];   // [m][k]
  __shared__ float Bs[2][32 * KS_S];    // [k][n]
  const int tid = threadIdx.x;
  const int lane = tid & 31, wave = tid >> 5;
  const int wm = wave & 3, wn = wave >> 2;
  const int lm = lane & 15, lh = lane >> 4;
  const int r0 = blockIdx.y * 128;
  const int c0 = blockIdx.x * 128;
  const int mA = tid >> 1, kA = (tid & 1) * 16;
  const int kB = tid >> 3, cB = (tid & 7) * 16;

  v8f acc[2][4] = {};

  auto stage = [&](int buf, int k0) {
    #pragma unroll
    for (int j = 0; j < 4; ++j)
      async_ld_b128(&X[(long)(r0 + mA) * K + k0 + kA + 4 * j],
                    &As[buf][mA * AS_F + kA + 4 * j]);
    #pragma unroll
    for (int j = 0; j < 4; ++j)
      async_ld_b128(&Th[(long)(k0 + kB) * co + c0 + cB + 4 * j],
                    &Bs[buf][kB * KS_S + cB + 4 * j]);
  };

  const int nt = K / 32;
  stage(0, 0);
  wait_async0();
  __syncthreads();

  for (int t = 0; t < nt; ++t) {
    const int cur = t & 1;
    if (t + 1 < nt) stage(cur ^ 1, (t + 1) * 32);
    const float* Ab = As[cur];
    const float* Bb = Bs[cur];
#if HAVE_WMMA_F32X4
    #pragma unroll
    for (int kk = 0; kk < 8; ++kk) {
      int kb = kk * 4 + lh * 2;
      v2f afr[2], bfr[4];
      #pragma unroll
      for (int tm = 0; tm < 2; ++tm)
        afr[tm] = *(const v2f*)&Ab[(wm * 32 + tm * 16 + lm) * AS_F + kb];
      #pragma unroll
      for (int tn = 0; tn < 4; ++tn) {
        int nc = wn * 64 + tn * 16 + lm;
        bfr[tn].x = Bb[kb * KS_S + nc];
        bfr[tn].y = Bb[(kb + 1) * KS_S + nc];
      }
      #pragma unroll
      for (int tm = 0; tm < 2; ++tm)
        #pragma unroll
        for (int tn = 0; tn < 4; ++tn)
          acc[tm][tn] = __builtin_amdgcn_wmma_f32_16x16x4_f32(
              false, afr[tm], false, bfr[tn], (short)0, acc[tm][tn], false, false);
    }
#else
    for (int kk2 = 0; kk2 < 32; ++kk2) {
      float bv[4];
      #pragma unroll
      for (int tn = 0; tn < 4; ++tn) bv[tn] = Bb[kk2 * KS_S + wn * 64 + tn * 16 + lm];
      #pragma unroll
      for (int tm = 0; tm < 2; ++tm)
        #pragma unroll
        for (int v = 0; v < 8; ++v) {
          float av = Ab[(wm * 32 + tm * 16 + lh * 8 + v) * AS_F + kk2];
          #pragma unroll
          for (int tn = 0; tn < 4; ++tn) acc[tm][tn][v] += av * bv[tn];
        }
    }
#endif
    wait_async0();
    __syncthreads();
  }
  #pragma unroll
  for (int tm = 0; tm < 2; ++tm)
    #pragma unroll
    for (int tn = 0; tn < 4; ++tn)
      #pragma unroll
      for (int v = 0; v < 8; ++v) {
        int i = r0 + wm * 32 + tm * 16 + lh * 8 + v;
        int c = c0 + wn * 64 + tn * 16 + lm;
        Y[(long)i * co + c] = dinv[i] * acc[tm][tn][v];
      }
}

// ---------------------------------------------------------------------------
// GEMM 2: Out[i][c] = relu( dinv[i] * sum_r A[r][i]*Ys[r][c] + bias[c] )
// (A includes self loops, so the +ys[i] term is implicit.)  fp32 WMMA 16x16x4.
// Async double-buffered LDS staging (A streams 64MB; stays L2-resident).
// ---------------------------------------------------------------------------
__global__ __launch_bounds__(256) void k_gemm_agg(const float* __restrict__ A,
                                                  const float* __restrict__ Ys,
                                                  const float* __restrict__ dinv,
                                                  const float* __restrict__ bias,
                                                  float* __restrict__ Out,
                                                  int n, int co) {
  __shared__ float As[2][32 * KS_S];   // [k][m] (direct-staged rows of A)
  __shared__ float Bs[2][32 * KS_S];   // [k][n]
  const int tid = threadIdx.x;
  const int lane = tid & 31, wave = tid >> 5;
  const int wm = wave & 3, wn = wave >> 2;
  const int lm = lane & 15, lh = lane >> 4;
  const int r0 = blockIdx.y * 128;
  const int c0 = blockIdx.x * 128;
  const int kA = tid >> 3, mA = (tid & 7) * 16;
  const int kB = tid >> 3, cB = (tid & 7) * 16;

  v8f acc[2][4] = {};

  auto stage = [&](int buf, int k0) {
    #pragma unroll
    for (int j = 0; j < 4; ++j)
      async_ld_b128(&A[(long)(k0 + kA) * n + r0 + mA + 4 * j],
                    &As[buf][kA * KS_S + mA + 4 * j]);
    #pragma unroll
    for (int j = 0; j < 4; ++j)
      async_ld_b128(&Ys[(long)(k0 + kB) * co + c0 + cB + 4 * j],
                    &Bs[buf][kB * KS_S + cB + 4 * j]);
  };

  const int nt = n / 32;
  stage(0, 0);
  wait_async0();
  __syncthreads();

  for (int t = 0; t < nt; ++t) {
    const int cur = t & 1;
    if (t + 1 < nt) stage(cur ^ 1, (t + 1) * 32);
    const float* Ab = As[cur];
    const float* Bb = Bs[cur];
#if HAVE_WMMA_F32X4
    #pragma unroll
    for (int kk = 0; kk < 8; ++kk) {
      int kb = kk * 4 + lh * 2;
      v2f afr[2], bfr[4];
      #pragma unroll
      for (int tm = 0; tm < 2; ++tm) {
        int mr = wm * 32 + tm * 16 + lm;
        afr[tm].x = Ab[kb * KS_S + mr];
        afr[tm].y = Ab[(kb + 1) * KS_S + mr];
      }
      #pragma unroll
      for (int tn = 0; tn < 4; ++tn) {
        int nc = wn * 64 + tn * 16 + lm;
        bfr[tn].x = Bb[kb * KS_S + nc];
        bfr[tn].y = Bb[(kb + 1) * KS_S + nc];
      }
      #pragma unroll
      for (int tm = 0; tm < 2; ++tm)
        #pragma unroll
        for (int tn = 0; tn < 4; ++tn)
          acc[tm][tn] = __builtin_amdgcn_wmma_f32_16x16x4_f32(
              false, afr[tm], false, bfr[tn], (short)0, acc[tm][tn], false, false);
    }
#else
    for (int kk2 = 0; kk2 < 32; ++kk2) {
      float bv[4];
      #pragma unroll
      for (int tn = 0; tn < 4; ++tn) bv[tn] = Bb[kk2 * KS_S + wn * 64 + tn * 16 + lm];
      #pragma unroll
      for (int tm = 0; tm < 2; ++tm)
        #pragma unroll
        for (int v = 0; v < 8; ++v) {
          float av = Ab[kk2 * KS_S + wm * 32 + tm * 16 + lh * 8 + v];
          #pragma unroll
          for (int tn = 0; tn < 4; ++tn) acc[tm][tn][v] += av * bv[tn];
        }
    }
#endif
    wait_async0();
    __syncthreads();
  }
  #pragma unroll
  for (int tm = 0; tm < 2; ++tm)
    #pragma unroll
    for (int tn = 0; tn < 4; ++tn)
      #pragma unroll
      for (int v = 0; v < 8; ++v) {
        int i = r0 + wm * 32 + tm * 16 + lh * 8 + v;
        int c = c0 + wn * 64 + tn * 16 + lm;
        float val = dinv[i] * acc[tm][tn][v] + bias[c];
        Out[(long)i * co + c] = val > 0.f ? val : 0.f;
      }
}

// ---------------------------------------------------------------------------
// GEMM 3: A2[i][j] = sum_k A[i][k]*A[k][j], diagonal zeroed.  A entries are
// small integers (exact in f16) -> v_wmma_f32_16x16x32_f16, convert-on-stage.
// ---------------------------------------------------------------------------
__global__ __launch_bounds__(256) void k_gemm_aug_f16(const float* __restrict__ A,
                                                      float* __restrict__ A2, int n) {
  __shared__ _Float16 Ah[128 * AH_S];  // [m][k]
  __shared__ _Float16 Bh[128 * AH_S];  // [nCol][k] (transposed on stage)
  const int tid = threadIdx.x;
  const int lane = tid & 31, wave = tid >> 5;
  const int wm = wave & 3, wn = wave >> 2;
  const int lm = lane & 15, lh = lane >> 4;
  const int r0 = blockIdx.y * 128;
  const int c0 = blockIdx.x * 128;
  const int mA = tid >> 1, kA = (tid & 1) * 16;
  const int kB = tid & 31, cB = (tid >> 5) * 16;

  v8f acc[2][4] = {};

  for (int k0 = 0; k0 < n; k0 += 32) {
    __syncthreads();
    if (k0 + 32 < n) {
      __builtin_prefetch(&A[(long)(r0 + mA) * n + k0 + 32 + kA], 0, 0);
      __builtin_prefetch(&A[(long)(k0 + 32 + kB) * n + c0 + cB], 0, 0);
    }
    #pragma unroll
    for (int j = 0; j < 4; ++j) {
      float4 v = *(const float4*)&A[(long)(r0 + mA) * n + k0 + kA + 4 * j];
      int b = mA * AH_S + kA + 4 * j;
      Ah[b + 0] = (_Float16)v.x; Ah[b + 1] = (_Float16)v.y;
      Ah[b + 2] = (_Float16)v.z; Ah[b + 3] = (_Float16)v.w;
    }
    #pragma unroll
    for (int j = 0; j < 4; ++j) {
      float4 v = *(const float4*)&A[(long)(k0 + kB) * n + c0 + cB + 4 * j];
      int cb = cB + 4 * j;
      Bh[(cb + 0) * AH_S + kB] = (_Float16)v.x;
      Bh[(cb + 1) * AH_S + kB] = (_Float16)v.y;
      Bh[(cb + 2) * AH_S + kB] = (_Float16)v.z;
      Bh[(cb + 3) * AH_S + kB] = (_Float16)v.w;
    }
    __syncthreads();
    v16h af[2], bf[4];
    #pragma unroll
    for (int tm = 0; tm < 2; ++tm) {
      int m = wm * 32 + tm * 16 + lm;
      #pragma unroll
      for (int v8 = 0; v8 < 8; ++v8) {
        int kv = ((v8 < 4) ? (2 * v8) : (16 + 2 * (v8 - 4))) + 8 * lh;
        v2h p = *(const v2h*)&Ah[m * AH_S + kv];
        af[tm][2 * v8] = p.x; af[tm][2 * v8 + 1] = p.y;
      }
    }
    #pragma unroll
    for (int tn = 0; tn < 4; ++tn) {
      int nc = wn * 64 + tn * 16 + lm;
      #pragma unroll
      for (int v8 = 0; v8 < 8; ++v8) {
        int kv = ((v8 < 4) ? (2 * v8) : (16 + 2 * (v8 - 4))) + 8 * lh;
        v2h p = *(const v2h*)&Bh[nc * AH_S + kv];
        bf[tn][2 * v8] = p.x; bf[tn][2 * v8 + 1] = p.y;
      }
    }
    #pragma unroll
    for (int tm = 0; tm < 2; ++tm)
      #pragma unroll
      for (int tn = 0; tn < 4; ++tn)
        acc[tm][tn] = __builtin_amdgcn_wmma_f32_16x16x32_f16(
            false, af[tm], false, bf[tn], (short)0, acc[tm][tn], false, false);
  }
  #pragma unroll
  for (int tm = 0; tm < 2; ++tm)
    #pragma unroll
    for (int tn = 0; tn < 4; ++tn)
      #pragma unroll
      for (int v = 0; v < 8; ++v) {
        int i = r0 + wm * 32 + tm * 16 + lh * 8 + v;
        int j = c0 + wn * 64 + tn * 16 + lm;
        A2[(long)i * n + j] = (i == j) ? 0.f : acc[tm][tn][v];
      }
}

// ---------------------------------------------------------------------------
// Top-k machinery
// ---------------------------------------------------------------------------
__global__ __launch_bounds__(256) void k_pnorm(const float* __restrict__ p, int len,
                                               float* __restrict__ out) {
  __shared__ float sh[256];
  float s = 0.f;
  for (int i = threadIdx.x; i < len; i += 256) s += p[i] * p[i];
  sh[threadIdx.x] = s;
  __syncthreads();
  for (int o = 128; o > 0; o >>= 1) {
    if (threadIdx.x < o) sh[threadIdx.x] += sh[threadIdx.x + o];
    __syncthreads();
  }
  if (threadIdx.x == 0) out[0] = sqrtf(sh[0]);
}

__global__ __launch_bounds__(256) void k_score(const float* __restrict__ x,
                                               const float* __restrict__ p, int ci,
                                               const float* __restrict__ pn,
                                               float* __restrict__ score, int n) {
  int wave = threadIdx.x >> 5, lane = threadIdx.x & 31;
  int i = blockIdx.x * 8 + wave;
  if (i >= n) return;
  float s = 0.f;
  for (int c = lane; c < ci; c += 32) s += x[(long)i * ci + c] * p[c];
  #pragma unroll
  for (int o = 16; o > 0; o >>= 1) s += __shfl_xor(s, o, 32);
  if (lane == 0) score[i] = tanhf(s / pn[0]);
}

#define SORTN 4096
__global__ __launch_bounds__(1024) void k_topk(const float* __restrict__ score, int n, int k,
                                               int* __restrict__ perm, float* __restrict__ vals) {
  __shared__ float ss[SORTN];
  __shared__ int si[SORTN];
  int tid = threadIdx.x;
  for (int i = tid; i < SORTN; i += 1024) {
    ss[i] = (i < n) ? score[i] : -3.402823e38f;
    si[i] = (i < n) ? i : 0x7fffffff;
  }
  __syncthreads();
  for (int sz = 2; sz <= SORTN; sz <<= 1) {
    for (int j = sz >> 1; j > 0; j >>= 1) {
      for (int t = tid; t < SORTN; t += 1024) {
        int ixj = t ^ j;
        if (ixj > t) {
          bool desc = ((t & sz) == 0);
          float sa = ss[t], sb = ss[ixj];
          int ia = si[t], ib = si[ixj];
          bool a_after = (sa < sb) || (sa == sb && ia > ib);
          bool b_after = (sb < sa) || (sb == sa && ib > ia);
          if (desc ? a_after : b_after) {
            ss[t] = sb; ss[ixj] = sa;
            si[t] = ib; si[ixj] = ia;
          }
        }
      }
      __syncthreads();
    }
  }
  for (int i = tid; i < k; i += 1024) { perm[i] = si[i]; vals[i] = ss[i]; }
}

__global__ __launch_bounds__(256) void k_pool_x(const float* __restrict__ x,
                                                const int* __restrict__ perm,
                                                const float* __restrict__ vals,
                                                float* __restrict__ xo, int co) {
  int j = blockIdx.x;
  float v = vals[j];
  int src = perm[j];
  for (int c = threadIdx.x; c < co; c += blockDim.x)
    xo[(long)j * co + c] = x[(long)src * co + c] * v;
}

// W_new[a][b] = A2[perm[a]][perm[b]] + (a==b)   (re-adds self loops for next level)
__global__ __launch_bounds__(256) void k_pool_adj(const float* __restrict__ A2,
                                                  const int* __restrict__ perm,
                                                  float* __restrict__ W, int nOld, int nNew) {
  int a = blockIdx.x;
  int ra = perm[a];
  const float* row = A2 + (long)ra * nOld;
  int b = blockIdx.y * blockDim.x + threadIdx.x;
  if (b < nNew) {
    float v = row[perm[b]];
    if (a == b) v += 1.0f;
    W[(long)a * nNew + b] = v;
  }
}

__global__ __launch_bounds__(256) void k_scatter_add(float* __restrict__ h,
                                                     const int* __restrict__ perm,
                                                     const float* __restrict__ add, int co) {
  int j = blockIdx.x;
  int dr = perm[j];
  for (int c = threadIdx.x; c < co; c += blockDim.x)
    h[(long)dr * co + c] += add[(long)j * co + c];
}

__global__ __launch_bounds__(256) void k_mean(const float* __restrict__ xb, int n, int co,
                                              float* __restrict__ out) {
  int c = blockIdx.x * blockDim.x + threadIdx.x;
  if (c < co) {
    float s = 0.f;
    for (int i = 0; i < n; ++i) s += xb[(long)i * co + c];
    out[c] = s / (float)n;
  }
}

// ---------------------------------------------------------------------------
// Host orchestration
// ---------------------------------------------------------------------------
extern "C" void kernel_launch(void* const* d_in, const int* in_sizes, int n_in,
                              void* d_out, int out_size, void* d_ws, size_t ws_size,
                              hipStream_t stream) {
  const float* x   = (const float*)d_in[0];
  const int*   ei  = (const int*)d_in[1];
  const float* th0 = (const float*)d_in[2];
  const float* b0  = (const float*)d_in[3];
  const float* th1 = (const float*)d_in[4];
  const float* b1  = (const float*)d_in[5];
  const float* th2 = (const float*)d_in[6];
  const float* b2  = (const float*)d_in[7];
  const float* th3 = (const float*)d_in[8];
  const float* b3  = (const float*)d_in[9];
  const float* th4 = (const float*)d_in[10];
  const float* b4  = (const float*)d_in[11];
  const float* p1  = (const float*)d_in[12];
  const float* p2  = (const float*)d_in[13];
  float* out = (float*)d_out;
  const int E = in_sizes[1] / 2;
  (void)n_in; (void)out_size; (void)ws_size;

  char* ws = (char*)d_ws;
  size_t off = 0;
  auto alloc = [&](size_t bytes) -> char* {
    char* p = ws + off;
    off += (bytes + 255) & ~(size_t)255;
    return p;
  };
  float* A0    = (float*)alloc((size_t)N0 * N0 * 4);   // level-0 A = W + I (persists)
  float* A2b   = (float*)alloc((size_t)N0 * N0 * 4);   // augment output (reused per level)
  float* A1    = (float*)alloc((size_t)N1 * N1 * 4);   // level-1 A (incl +I)
  float* A2lvl = (float*)alloc((size_t)N2 * N2 * 4);   // level-2 A (incl +I)
  float* dinv0 = (float*)alloc(N0 * 4);
  float* dinv1 = (float*)alloc(N1 * 4);
  float* dinv2 = (float*)alloc(N2 * 4);
  float* ybuf  = (float*)alloc((size_t)N0 * 512 * 4);  // shared feature-GEMM output
  float* x0    = (float*)alloc((size_t)N0 * 256 * 4);
  float* x1    = (float*)alloc((size_t)N1 * 256 * 4);
  float* x1b   = (float*)alloc((size_t)N1 * 512 * 4);
  float* x2    = (float*)alloc((size_t)N2 * 512 * 4);
  float* xb    = (float*)alloc((size_t)N2 * 512 * 4);
  float* xu    = (float*)alloc((size_t)N1 * 256 * 4);
  float* h1    = (float*)alloc((size_t)N1 * 512 * 4);
  float* h0    = (float*)alloc((size_t)N0 * 256 * 4);
  float* score = (float*)alloc(N0 * 4);
  int*   perm1 = (int*)alloc(N1 * 4);
  float* vals1 = (float*)alloc(N1 * 4);
  int*   perm2 = (int*)alloc(N2 * 4);
  float* vals2 = (float*)alloc(N2 * 4);
  float* pn    = (float*)alloc(2 * 4);

  // ---- Build A0 = W + I ----
  hipMemsetAsync(A0, 0, (size_t)N0 * N0 * 4, stream);
  k_scatter_edges<<<(E + 255) / 256, 256, 0, stream>>>(ei, A0, N0, E);
  k_add_eye<<<(N0 + 255) / 256, 256, 0, stream>>>(A0, N0);
  k_row_dinv<<<N0 / 8, 256, 0, stream>>>(A0, dinv0, N0);

  // ---- Down block 0: x0 = relu(gcn(x, W, th0, b0))  [4096,256] ----
  k_gemm_feat<<<dim3(256 / 128, N0 / 128), 256, 0, stream>>>(x, th0, dinv0, ybuf, N0, 128, 256);
  k_gemm_agg<<<dim3(256 / 128, N0 / 128), 256, 0, stream>>>(A0, ybuf, dinv0, b0, x0, N0, 256);

  // ---- augment_adj(W) -> A2b, then top-k pool (k = 2048) ----
  k_gemm_aug_f16<<<dim3(N0 / 128, N0 / 128), 256, 0, stream>>>(A0, A2b, N0);
  k_pnorm<<<1, 256, 0, stream>>>(p1, 256, pn);
  k_score<<<N0 / 8, 256, 0, stream>>>(x0, p1, 256, pn, score, N0);
  k_topk<<<1, 1024, 0, stream>>>(score, N0, N1, perm1, vals1);
  k_pool_x<<<N1, 256, 0, stream>>>(x0, perm1, vals1, x1, 256);
  k_pool_adj<<<dim3(N1, N1 / 256), 256, 0, stream>>>(A2b, perm1, A1, N0, N1);
  k_row_dinv<<<N1 / 8, 256, 0, stream>>>(A1, dinv1, N1);

  // ---- Down block 1: x1b = relu(gcn(x1, W1, th1, b1))  [2048,512] ----
  k_gemm_feat<<<dim3(512 / 128, N1 / 128), 256, 0, stream>>>(x1, th1, dinv1, ybuf, N1, 256, 512);
  k_gemm_agg<<<dim3(512 / 128, N1 / 128), 256, 0, stream>>>(A1, ybuf, dinv1, b1, x1b, N1, 512);

  // ---- augment_adj(W1) -> A2b, top-k pool (k = 1024) ----
  k_gemm_aug_f16<<<dim3(N1 / 128, N1 / 128), 256, 0, stream>>>(A1, A2b, N1);
  k_pnorm<<<1, 256, 0, stream>>>(p2, 512, pn);
  k_score<<<N1 / 8, 256, 0, stream>>>(x1b, p2, 512, pn, score, N1);
  k_topk<<<1, 1024, 0, stream>>>(score, N1, N2, perm2, vals2);
  k_pool_x<<<N2, 256, 0, stream>>>(x1b, perm2, vals2, x2, 512);
  k_pool_adj<<<dim3(N2, N2 / 256), 256, 0, stream>>>(A2b, perm2, A2lvl, N1, N2);
  k_row_dinv<<<N2 / 8, 256, 0, stream>>>(A2lvl, dinv2, N2);

  // ---- Bottleneck: xb = relu(gcn(x2, W2, th2, b2))  [1024,512]; summary ----
  k_gemm_feat<<<dim3(512 / 128, N2 / 128), 256, 0, stream>>>(x2, th2, dinv2, ybuf, N2, 512, 512);
  k_gemm_agg<<<dim3(512 / 128, N2 / 128), 256, 0, stream>>>(A2lvl, ybuf, dinv2, b2, xb, N2, 512);
  k_mean<<<2, 256, 0, stream>>>(xb, N2, 512, out + (size_t)N0 * 128);

  // ---- Up block (level 1): xu = relu(gcn(x1b + up(xb), W1, th3, b3)) [2048,256] ----
  hipMemcpyAsync(h1, x1b, (size_t)N1 * 512 * 4, hipMemcpyDeviceToDevice, stream);
  k_scatter_add<<<N2, 256, 0, stream>>>(h1, perm2, xb, 512);
  k_gemm_feat<<<dim3(256 / 128, N1 / 128), 256, 0, stream>>>(h1, th3, dinv1, ybuf, N1, 512, 256);
  k_gemm_agg<<<dim3(256 / 128, N1 / 128), 256, 0, stream>>>(A1, ybuf, dinv1, b3, xu, N1, 256);

  // ---- Up block (level 0): xout = relu(gcn(x0 + up(xu), W, th4, b4)) [4096,128] ----
  hipMemcpyAsync(h0, x0, (size_t)N0 * 256 * 4, hipMemcpyDeviceToDevice, stream);
  k_scatter_add<<<N1, 256, 0, stream>>>(h0, perm1, xu, 256);
  k_gemm_feat<<<dim3(128 / 128, N0 / 128), 256, 0, stream>>>(h0, th4, dinv0, ybuf, N0, 256, 128);
  k_gemm_agg<<<dim3(128 / 128, N0 / 128), 256, 0, stream>>>(A0, ybuf, dinv0, b4, out, N0, 128);
}
